// HierarchicalTransformerAttention_17643725652006
// MI455X (gfx1250) — compile-verified, matrix operand order checked by reading, not run on hardware
//
#include <hip/hip_runtime.h>
#include <hip/hip_bf16.h>

// ---------------------------------------------------------------------------
// Problem constants (fixed by the reference)
// ---------------------------------------------------------------------------
constexpr int Bc = 4;      // batch
constexpr int Sc = 1024;   // sequence
constexpr int Dc = 1024;   // model dim
constexpr int Hc = 16;     // heads
constexpr int DHc = 64;    // head dim
constexpr int Mc = 896;    // max_head_num (gathered rows)
constexpr int BMc = Bc * Mc;   // 3584 rows through the dense pipeline

typedef __attribute__((ext_vector_type(16))) _Float16 v16h;
typedef __attribute__((ext_vector_type(8)))  _Float16 v8h;
typedef __attribute__((ext_vector_type(4)))  _Float16 v4h;
typedef __attribute__((ext_vector_type(8)))  float    v8f;

// ---------------------------------------------------------------------------
// WMMA helpers
// ---------------------------------------------------------------------------
__device__ __forceinline__ v8f wmma_f16(v16h a, v16h b, v8f c) {
    return __builtin_amdgcn_wmma_f32_16x16x32_f16(
        /*neg_a=*/false, a, /*neg_b=*/false, b,
        /*c_mod=*/(short)0, c, /*reuse_a=*/false, /*reuse_b=*/false);
}

// Assemble a 16x32 f16 operand fragment from a lane-resolved pointer.
// CDNA5 A-operand layout: lane L (L<16) holds row L, K = {hl*8+0..7, 16+hl*8+0..7}
// with hl = lane>>4.  The caller bakes (row+lane15)*ld + hl*8 into `p`, so
// every fragment is exactly two 16-byte loads at constant displacements —
// the compiler keeps ONE address vector per matrix and folds the rest into
// the global_load immediate offset.
__device__ __forceinline__ v16h load_frag_ptr(const _Float16* p) {
    v8h lo = *(const v8h*)(p);        // K = k0 + hl*8 + 0..7
    v8h hi = *(const v8h*)(p + 16);   // K = k0 + 16 + hl*8 + 0..7
    v16h out;
#pragma unroll
    for (int i = 0; i < 8; ++i) { out[i] = lo[i]; out[i + 8] = hi[i]; }
    return out;
}

// ---------------------------------------------------------------------------
// 1. Stable partition: valid (level != 0) tokens first.  One block per batch,
//    1024-thread Hillis-Steele scan in LDS.
// ---------------------------------------------------------------------------
__global__ void hta_build_order(const int* __restrict__ lh,
                                int* __restrict__ order,
                                int* __restrict__ invord,
                                int* __restrict__ lr_g) {
    __shared__ int sv[Sc];
    const int b = blockIdx.x, i = threadIdx.x;
    const int lev = lh[b * Sc + i];
    const int valid = (lev != 0) ? 1 : 0;
    sv[i] = valid;
    __syncthreads();
    for (int off = 1; off < Sc; off <<= 1) {
        int t = (i >= off) ? sv[i - off] : 0;
        __syncthreads();
        sv[i] += t;
        __syncthreads();
    }
    const int cnt = sv[Sc - 1];                 // # valid tokens in batch
    const int dest = valid ? (sv[i] - 1) : (cnt + i - sv[i]);
    order[b * Sc + dest] = i;
    invord[b * Sc + i]   = dest;
    if (dest < Mc) lr_g[b * Mc + dest] = lev;   // gathered levels (0 => masked)
}

// ---------------------------------------------------------------------------
// 2. Gather the first M reordered rows; keep f32 (residual) + f16 (WMMA A).
// ---------------------------------------------------------------------------
__global__ void hta_gather_rows(const float* __restrict__ hs,
                                const int* __restrict__ order,
                                float* __restrict__ hr32,
                                _Float16* __restrict__ hr16) {
    const int b = blockIdx.x / Mc, m = blockIdx.x % Mc;
    const int src = order[b * Sc + m];
    const float4 v = ((const float4*)(hs + ((size_t)b * Sc + src) * Dc))[threadIdx.x];
    ((float4*)(hr32 + ((size_t)b * Mc + m) * Dc))[threadIdx.x] = v;
    v4h h; h.x = (_Float16)v.x; h.y = (_Float16)v.y; h.z = (_Float16)v.z; h.w = (_Float16)v.w;
    ((v4h*)(hr16 + ((size_t)b * Mc + m) * Dc))[threadIdx.x] = h;
}

// ---------------------------------------------------------------------------
// 3. Weight convert+transpose: W (D x N, f32 row-major) -> WT (N x D, f16).
//    Classic 32x32 LDS tile transpose so both sides stay coalesced.
// ---------------------------------------------------------------------------
__global__ void hta_transpose_w(const float* __restrict__ W, _Float16* __restrict__ WT,
                                int Drows, int Ncols) {
    __shared__ float tile[32][33];
    const int n0 = blockIdx.x * 32, d0 = blockIdx.y * 32;
    const int tx = threadIdx.x, ty = threadIdx.y;
#pragma unroll
    for (int r = 0; r < 32; r += 8)
        tile[ty + r][tx] = W[(size_t)(d0 + ty + r) * Ncols + n0 + tx];
    __syncthreads();
#pragma unroll
    for (int r = 0; r < 32; r += 8)
        WT[(size_t)(n0 + ty + r) * Drows + d0 + tx] = (_Float16)tile[tx][ty + r];
}

// ---------------------------------------------------------------------------
// 4. WMMA GEMM:  C[Mr x N] = A[Mr x K] @ BT^T + bias.   A,BT f16 row-major
//    (BT is the transposed weight, N x K).  Block = 128x128, 8 waves, each
//    wave owns a 32x64 patch = 2x4 accumulators; K-step 32.
//    Addressing: ONE lane-resolved base pointer for A and one for BT; all
//    fragment variants (row+16, col+c*16, k0) are constant displacements
//    (max 3*16*K*2 = 96 KB, well inside the 24-bit global_load offset).
// ---------------------------------------------------------------------------
__global__ void hta_gemm_wmma(const _Float16* __restrict__ A,
                              const _Float16* __restrict__ BT,
                              const float* __restrict__ bias,
                              float* __restrict__ Cf, _Float16* __restrict__ Ch,
                              int Mr, int N, int K) {
    const int lane = threadIdx.x & 31;
    const int wave = threadIdx.x >> 5;
    const int hl = lane >> 4, col = lane & 15;
    const int row0 = blockIdx.x * 128 + (wave & 3) * 32;
    const int col0 = blockIdx.y * 128 + (wave >> 2) * 64;

    const _Float16* pA = A  + (size_t)(row0 + col) * K + hl * 8;
    const _Float16* pB = BT + (size_t)(col0 + col) * K + hl * 8;
    const size_t rK = (size_t)16 * K;   // one 16-row tile, in elements

    const v8f z = {0.f, 0.f, 0.f, 0.f, 0.f, 0.f, 0.f, 0.f};
    v8f acc[2][4];
#pragma unroll
    for (int r = 0; r < 2; ++r)
#pragma unroll
        for (int c = 0; c < 4; ++c) acc[r][c] = z;

    for (int k0 = 0; k0 < K; k0 += 32) {
        __builtin_prefetch(pA + k0 + 64, 0, 1);      // global_prefetch_b8
        v16h a0 = load_frag_ptr(pA + k0);
        v16h a1 = load_frag_ptr(pA + rK + k0);
        v16h b0 = load_frag_ptr(pB + k0);
        v16h b1 = load_frag_ptr(pB + rK + k0);
        v16h b2 = load_frag_ptr(pB + 2 * rK + k0);
        v16h b3 = load_frag_ptr(pB + 3 * rK + k0);
        acc[0][0] = wmma_f16(a0, b0, acc[0][0]);
        acc[0][1] = wmma_f16(a0, b1, acc[0][1]);
        acc[0][2] = wmma_f16(a0, b2, acc[0][2]);
        acc[0][3] = wmma_f16(a0, b3, acc[0][3]);
        acc[1][0] = wmma_f16(a1, b0, acc[1][0]);
        acc[1][1] = wmma_f16(a1, b1, acc[1][1]);
        acc[1][2] = wmma_f16(a1, b2, acc[1][2]);
        acc[1][3] = wmma_f16(a1, b3, acc[1][3]);
    }

#pragma unroll
    for (int r = 0; r < 2; ++r)
#pragma unroll
        for (int c = 0; c < 4; ++c) {
            const int gcol = col0 + c * 16 + col;
            const float bb = bias[gcol];
#pragma unroll
            for (int e = 0; e < 8; ++e) {
                const int grow = row0 + r * 16 + e + 8 * hl;
                const float v = acc[r][c][e] + bb;
                const size_t idx = (size_t)grow * N + gcol;
                if (Cf) Cf[idx] = v;
                if (Ch) Ch[idx] = (_Float16)v;
            }
        }
}

// ---------------------------------------------------------------------------
// 5. V transpose: v (B, M, H*DH) f16 -> vT (B, H, DH, M) f16 so that P@V
//    B-operand fragments are contiguous along the key dimension.
// ---------------------------------------------------------------------------
__global__ void hta_v_transpose(const _Float16* __restrict__ vh, _Float16* __restrict__ vT) {
    __shared__ _Float16 tile[32][34];
    const int b = blockIdx.z / Hc, h = blockIdx.z % Hc;
    const int m0 = blockIdx.x * 32, d0 = blockIdx.y * 32;
    const int tx = threadIdx.x, ty = threadIdx.y;
#pragma unroll
    for (int r = 0; r < 32; r += 8)
        tile[ty + r][tx] = vh[((size_t)(b * Mc + m0 + ty + r)) * Dc + h * DHc + d0 + tx];
    __syncthreads();
#pragma unroll
    for (int r = 0; r < 32; r += 8)
        vT[((size_t)((b * Hc + h) * DHc + d0 + ty + r)) * Mc + m0 + tx] = tile[tx][ty + r];
}

// ---------------------------------------------------------------------------
// 6. Attention, flash style.  One wave owns a 16-query tile of one (b,h);
//    28 kv-steps of 32 keys.  Score WMMAs -> bias/mask -> online softmax
//    (shfl_xor reductions confined to 16-lane halves, wave32) -> P relayout
//    through wave-private LDS -> P@V WMMAs.  One base pointer per matrix;
//    kv/t/dt displacements are immediates (max ~90 KB).
// ---------------------------------------------------------------------------
__global__ void hta_attention(const _Float16* __restrict__ qh,
                              const _Float16* __restrict__ kh,
                              const _Float16* __restrict__ vT,
                              const int* __restrict__ lr_g,
                              const float* __restrict__ bias_table,
                              _Float16* __restrict__ attn_h) {
    __shared__ __align__(16) _Float16 pbuf[8][16 * 32];   // 8 waves x 1 KB
    const int lane = threadIdx.x & 31;
    const int wave = threadIdx.x >> 5;
    const int hl = lane >> 4, col = lane & 15;
    const int b = blockIdx.y / Hc, h = blockIdx.y % Hc;
    const int q0 = (blockIdx.x * 8 + wave) * 16;
    const int rb = b * Mc;

    // Lane-resolved base pointers.
    const _Float16* pQ = qh + (size_t)(rb + q0 + col) * Dc + h * DHc + hl * 8;
    const _Float16* pK = kh + (size_t)(rb + col) * Dc + h * DHc + hl * 8;
    const _Float16* pV = vT + (size_t)((b * Hc + h) * DHc + col) * Mc + hl * 8;

    // Q fragments (constant over the kv loop): A operand, K dim = head dim.
    v16h qf[2];
    qf[0] = load_frag_ptr(pQ);
    qf[1] = load_frag_ptr(pQ + 32);

    int lrq[8];
#pragma unroll
    for (int e = 0; e < 8; ++e) lrq[e] = lr_g[rb + q0 + e + 8 * hl];

    float rmax[8], rsum[8];
#pragma unroll
    for (int e = 0; e < 8; ++e) { rmax[e] = -1e30f; rsum[e] = 0.f; }

    const v8f z = {0.f, 0.f, 0.f, 0.f, 0.f, 0.f, 0.f, 0.f};
    v8f o[4] = {z, z, z, z};

    for (int kv = 0; kv < Mc; kv += 32) {
        // ---- scores: S (16q x 32kv) = Q @ K^T, two 16-col tiles -----------
        v8f sa[2] = {z, z};
#pragma unroll
        for (int t = 0; t < 2; ++t) {
            const _Float16* pk = pK + (size_t)(kv + t * 16) * Dc;
            v16h kf0 = load_frag_ptr(pk);
            v16h kf1 = load_frag_ptr(pk + 32);
            sa[t] = wmma_f16(qf[0], kf0, sa[t]);
            sa[t] = wmma_f16(qf[1], kf1, sa[t]);
        }

        // ---- relative-level bias + mask -----------------------------------
        float p[2][8];
        float mrow[8];
#pragma unroll
        for (int e = 0; e < 8; ++e) mrow[e] = -1e30f;
#pragma unroll
        for (int t = 0; t < 2; ++t) {
            const int lk = lr_g[rb + kv + t * 16 + col];
#pragma unroll
            for (int e = 0; e < 8; ++e) {
                float x = sa[t][e] * 0.125f;               // 1/sqrt(64)
                const bool valid = (lk != 0) && (lrq[e] != 0);
                int rel = valid ? (lk - lrq[e] + 6) : 0;
                rel = rel < 0 ? 0 : rel;
                x += bias_table[rel * Hc + h];
                if (!valid) x -= 10000.0f;
                p[t][e] = x;
                mrow[e] = fmaxf(mrow[e], x);
            }
        }

        // ---- online softmax (row reductions across the 16-lane half) ------
#pragma unroll
        for (int e = 0; e < 8; ++e)
#pragma unroll
            for (int off = 1; off < 16; off <<= 1)
                mrow[e] = fmaxf(mrow[e], __shfl_xor(mrow[e], off, 32));

        float scale[8];
#pragma unroll
        for (int e = 0; e < 8; ++e) {
            const float nm = fmaxf(rmax[e], mrow[e]);
            scale[e] = __expf(rmax[e] - nm);
            rmax[e] = nm;
            float s = 0.f;
#pragma unroll
            for (int t = 0; t < 2; ++t) { p[t][e] = __expf(p[t][e] - nm); s += p[t][e]; }
#pragma unroll
            for (int off = 1; off < 16; off <<= 1) s += __shfl_xor(s, off, 32);
            rsum[e] = rsum[e] * scale[e] + s;
        }
#pragma unroll
        for (int dt = 0; dt < 4; ++dt)
#pragma unroll
            for (int e = 0; e < 8; ++e) o[dt][e] *= scale[e];

        // ---- P: C-layout regs -> LDS -> A-layout fragment -----------------
#pragma unroll
        for (int t = 0; t < 2; ++t)
#pragma unroll
            for (int e = 0; e < 8; ++e)
                pbuf[wave][(e + 8 * hl) * 32 + t * 16 + col] = (_Float16)p[t][e];
        // wave-private region: same-wave DS ordering suffices (DScnt, in order)
        const _Float16* pb = &pbuf[wave][col * 32 + hl * 8];
        v8h plo = *(const v8h*)pb;
        v8h phi = *(const v8h*)(pb + 16);
        v16h pf;
#pragma unroll
        for (int i = 0; i < 8; ++i) { pf[i] = plo[i]; pf[i + 8] = phi[i]; }

        // ---- O += P @ V  (B operand from transposed V, contiguous in kv) --
#pragma unroll
        for (int dt = 0; dt < 4; ++dt) {
            v16h vf = load_frag_ptr(pV + (size_t)dt * 16 * Mc + kv);
            o[dt] = wmma_f16(pf, vf, o[dt]);
        }
    }

    // ---- normalize and store (B, M, H*DH) f16 for the Wo GEMM -------------
#pragma unroll
    for (int dt = 0; dt < 4; ++dt)
#pragma unroll
        for (int e = 0; e < 8; ++e) {
            const int grow = q0 + e + 8 * hl;
            const float vv = o[dt][e] / rsum[e];
            attn_h[((size_t)rb + grow) * Dc + h * DHc + dt * 16 + col] = (_Float16)vv;
        }
}

// ---------------------------------------------------------------------------
// 7. Residual + LayerNorm over D=1024, one block per row, in place on y.
// ---------------------------------------------------------------------------
__global__ void hta_residual_ln(float* __restrict__ y, const float* __restrict__ hr,
                                const float* __restrict__ gamma, const float* __restrict__ beta) {
    __shared__ float red[256];
    const int row = blockIdx.x, tid = threadIdx.x;
    float* yrow = y + (size_t)row * Dc;
    const float* hrow = hr + (size_t)row * Dc;
    float local[4];
    float s = 0.f;
#pragma unroll
    for (int i = 0; i < 4; ++i) {
        const int idx = tid + i * 256;
        const float v = yrow[idx] + hrow[idx];
        local[i] = v; s += v;
    }
    red[tid] = s; __syncthreads();
    for (int off = 128; off > 0; off >>= 1) { if (tid < off) red[tid] += red[tid + off]; __syncthreads(); }
    const float mu = red[0] / (float)Dc;
    __syncthreads();
    float s2 = 0.f;
#pragma unroll
    for (int i = 0; i < 4; ++i) { const float d = local[i] - mu; s2 += d * d; }
    red[tid] = s2; __syncthreads();
    for (int off = 128; off > 0; off >>= 1) { if (tid < off) red[tid] += red[tid + off]; __syncthreads(); }
    const float inv = rsqrtf(red[0] / (float)Dc + 1e-5f);
#pragma unroll
    for (int i = 0; i < 4; ++i) {
        const int idx = tid + i * 256;
        yrow[idx] = (local[i] - mu) * inv * gamma[idx] + beta[idx];
    }
}

// ---------------------------------------------------------------------------
// 8. Scatter back to original order; pass-through rows copied from input.
// ---------------------------------------------------------------------------
__global__ void hta_scatter(const float* __restrict__ y, const float* __restrict__ hs,
                            const int* __restrict__ invord, float* __restrict__ out) {
    const int b = blockIdx.y, j = blockIdx.x;
    const int dest = invord[b * Sc + j];
    const float4* src = (dest < Mc)
        ? (const float4*)(y  + ((size_t)b * Mc + dest) * Dc)
        : (const float4*)(hs + ((size_t)b * Sc + j)    * Dc);
    ((float4*)(out + ((size_t)b * Sc + j) * Dc))[threadIdx.x] = src[threadIdx.x];
}

// ---------------------------------------------------------------------------
// Launcher
// ---------------------------------------------------------------------------
extern "C" void kernel_launch(void* const* d_in, const int* in_sizes, int n_in,
                              void* d_out, int out_size, void* d_ws, size_t ws_size,
                              hipStream_t stream) {
    const float* hs     = (const float*)d_in[0];
    const float* Wq     = (const float*)d_in[1];
    const float* bq     = (const float*)d_in[2];
    const float* Wk     = (const float*)d_in[3];
    const float* bk     = (const float*)d_in[4];
    const float* Wv     = (const float*)d_in[5];
    const float* bv     = (const float*)d_in[6];
    const float* Wo     = (const float*)d_in[7];
    const float* bo     = (const float*)d_in[8];
    const float* btab   = (const float*)d_in[9];
    const float* gamma  = (const float*)d_in[10];
    const float* beta   = (const float*)d_in[11];
    const int*   lh     = (const int*)d_in[12];
    float* out = (float*)d_out;

    // ---- workspace layout (256-byte aligned slices) ----
    char* base = (char*)d_ws;
    size_t off = 0;
    auto alloc = [&](size_t bytes) -> void* {
        void* p = base + off;
        off += (bytes + 255) & ~(size_t)255;
        return p;
    };
    int*       order  = (int*)alloc((size_t)Bc * Sc * 4);
    int*       invord = (int*)alloc((size_t)Bc * Sc * 4);
    int*       lr_g   = (int*)alloc((size_t)Bc * Mc * 4);
    float*     hr32   = (float*)alloc((size_t)BMc * Dc * 4);
    float*     y      = (float*)alloc((size_t)BMc * Dc * 4);
    _Float16*  hr16   = (_Float16*)alloc((size_t)BMc * Dc * 2);
    _Float16*  WqT    = (_Float16*)alloc((size_t)Dc * Dc * 2);
    _Float16*  WkT    = (_Float16*)alloc((size_t)Dc * Dc * 2);
    _Float16*  WvT    = (_Float16*)alloc((size_t)Dc * Dc * 2);
    _Float16*  WoT    = (_Float16*)alloc((size_t)Dc * Dc * 2);
    _Float16*  qh     = (_Float16*)alloc((size_t)BMc * Dc * 2);
    _Float16*  kh     = (_Float16*)alloc((size_t)BMc * Dc * 2);
    _Float16*  vh     = (_Float16*)alloc((size_t)BMc * Dc * 2);
    _Float16*  vT     = (_Float16*)alloc((size_t)BMc * Dc * 2);
    _Float16*  attnh  = (_Float16*)alloc((size_t)BMc * Dc * 2);

    // 1. ordering / levels
    hta_build_order<<<Bc, Sc, 0, stream>>>(lh, order, invord, lr_g);
    // 2. gather + f16 convert
    hta_gather_rows<<<Bc * Mc, 256, 0, stream>>>(hs, order, hr32, hr16);
    // 3. weight transposes (f32 -> f16, N x D)
    {
        dim3 g(Dc / 32, Dc / 32), t(32, 8);
        hta_transpose_w<<<g, t, 0, stream>>>(Wq, WqT, Dc, Dc);
        hta_transpose_w<<<g, t, 0, stream>>>(Wk, WkT, Dc, Dc);
        hta_transpose_w<<<g, t, 0, stream>>>(Wv, WvT, Dc, Dc);
        hta_transpose_w<<<g, t, 0, stream>>>(Wo, WoT, Dc, Dc);
    }
    // 4. QKV projections (WMMA)
    {
        dim3 g(BMc / 128, Dc / 128);
        hta_gemm_wmma<<<g, 256, 0, stream>>>(hr16, WqT, bq, (float*)nullptr, qh, BMc, Dc, Dc);
        hta_gemm_wmma<<<g, 256, 0, stream>>>(hr16, WkT, bk, (float*)nullptr, kh, BMc, Dc, Dc);
        hta_gemm_wmma<<<g, 256, 0, stream>>>(hr16, WvT, bv, (float*)nullptr, vh, BMc, Dc, Dc);
    }
    // 5. V transpose to (B, H, DH, M)
    hta_v_transpose<<<dim3(Mc / 32, DHc / 32, Bc * Hc), dim3(32, 8), 0, stream>>>(vh, vT);
    // 6. attention
    hta_attention<<<dim3(Mc / (16 * 8), Bc * Hc), 256, 0, stream>>>(qh, kh, vT, lr_g, btab, attnh);
    // 7. output projection (WMMA, f32 out) + residual LayerNorm
    hta_gemm_wmma<<<dim3(BMc / 128, Dc / 128), 256, 0, stream>>>(attnh, WoT, bo, y, (_Float16*)nullptr, BMc, Dc, Dc);
    hta_residual_ln<<<BMc, 256, 0, stream>>>(y, hr32, gamma, beta);
    // 8. scatter back to original token order
    hta_scatter<<<dim3(Sc, Bc), 256, 0, stream>>>(y, hs, invord, out);
}
